// Longformer_25864293057124
// MI455X (gfx1250) — compile-verified, matrix operand order checked by reading, not run on hardware
//
#include <hip/hip_runtime.h>
#include <hip/hip_bf16.h>

#define B_  4
#define S_  2040
#define D_  768
#define H_  12
#define L_  6
#define FF_ 3072
#define W_  170
#define DH_ 64
#define C_  12          // S/W
#define M_  (B_*S_)     // 8160

#define KSTEP 64                 // K halves staged per LDS buffer
#define APAD  8                  // pad halves per row (row stride 144B: 16B-aligned, bank-friendly)
#define AROW  (KSTEP + APAD)     // 72 halves

typedef __attribute__((ext_vector_type(16))) _Float16 v16h;
typedef __attribute__((ext_vector_type(8)))  _Float16 v8h;
typedef __attribute__((ext_vector_type(8)))  float    v8f;

// ---------------- weight convert: W[K][N] f32 -> WT[N][K] f16 ----------------
__global__ void __launch_bounds__(256) cvt_wT_k(const float* __restrict__ Wsrc,
                                                _Float16* __restrict__ WT, int K, int N) {
  int idx = blockIdx.x * 256 + threadIdx.x;
  if (idx >= K * N) return;
  int n = idx / K, k = idx - n * K;
  WT[(size_t)n * K + k] = (_Float16)Wsrc[(size_t)k * N + n];
}

// ---------------- f32 -> f16 elementwise ----------------
__global__ void __launch_bounds__(256) cvt_h_k(const float* __restrict__ in,
                                               _Float16* __restrict__ out, int n) {
  int idx = blockIdx.x * 256 + threadIdx.x;
  if (idx < n) out[idx] = (_Float16)in[idx];
}

// ---------------- WMMA GEMM with async global->LDS A-tile staging ----------------
// X row-major f16 [M][K]; WT = W^T row-major f16 [N][K].
// Block = 8 waves: 64 M rows x 128 N cols. A tile (shared by all waves) is DMA'd to
// LDS with global_load_async_to_lds_b128, double-buffered on ASYNCcnt.
// mode 0: f32 out (+bias). mode 1: gelu(+bias) -> f16 out.
__global__ void __launch_bounds__(256) gemm_wmma_k(const _Float16* __restrict__ X,
                                                   const _Float16* __restrict__ WT,
                                                   const float* __restrict__ bias,
                                                   float* __restrict__ outF,
                                                   _Float16* __restrict__ outH,
                                                   int M, int N, int K, int mode) {
  __shared__ __align__(16) _Float16 sA[2][64 * AROW];   // 2 x 9216 B

  const int tid  = threadIdx.x;
  const int lane = tid & 31;
  const int wv   = tid >> 5;             // 8 waves
  const int g    = lane >> 4;            // lane group (K split)
  const int r    = lane & 15;            // M (A) / N (B) index within tile
  const int mBase = blockIdx.x * 64;
  const int col   = blockIdx.y * 128 + wv * 16 + r;

  // --- staging map: 512 x 16B chunks per tile, 2 per thread ---
  const int row0 = tid >> 3;                   // rows 0..31
  const int row1 = (tid + 256) >> 3;           // rows 32..63
  const int c8x  = (tid & 7) * 8;              // halves within row (16B units)
  int gr0 = mBase + row0; if (gr0 >= M) gr0 = M - 1;   // clamp; stores guarded later
  int gr1 = mBase + row1; if (gr1 >= M) gr1 = M - 1;
  const _Float16* gA0 = X + (size_t)gr0 * K + c8x;
  const _Float16* gA1 = X + (size_t)gr1 * K + c8x;
  // LDS byte address = low 32 bits of generic pointer (ISA: LDS logical addr = addr[31:0])
  const unsigned ldsBase = (unsigned)(unsigned long long)(uintptr_t)(&sA[0][0]);
  const unsigned lds0 = ldsBase + (unsigned)(row0 * AROW + c8x) * 2u;
  const unsigned lds1 = ldsBase + (unsigned)(row1 * AROW + c8x) * 2u;
  const unsigned bufBytes = (unsigned)(64 * AROW * 2);

  v8f acc[4] = {v8f{}, v8f{}, v8f{}, v8f{}};
  union F16x16 { v16h v; v8h h[2]; };
  const _Float16* wp = WT + (size_t)col * K;

  // ---- prologue: stage first tile ----
  asm volatile("global_load_async_to_lds_b128 %0, %1, off"
               :: "v"(lds0), "v"(gA0) : "memory");
  asm volatile("global_load_async_to_lds_b128 %0, %1, off"
               :: "v"(lds1), "v"(gA1) : "memory");

  const int nstage = K / KSTEP;
  for (int st = 0; st < nstage; ++st) {
    const int kk = st * KSTEP;
    if (st + 1 < nstage) {
      const unsigned boff = ((st + 1) & 1) ? bufBytes : 0u;
      asm volatile("global_load_async_to_lds_b128 %0, %1, off"
                   :: "v"(lds0 + boff), "v"(gA0 + kk + KSTEP) : "memory");
      asm volatile("global_load_async_to_lds_b128 %0, %1, off"
                   :: "v"(lds1 + boff), "v"(gA1 + kk + KSTEP) : "memory");
      asm volatile("s_wait_asynccnt 0x2" ::: "memory");   // previous stage done (in-order)
    } else {
      asm volatile("s_wait_asynccnt 0x0" ::: "memory");
    }
    __syncthreads();                                       // tile visible to all waves

    const _Float16* abuf = &sA[st & 1][0];
#pragma unroll
    for (int ks = 0; ks < KSTEP; ks += 32) {
      F16x16 bfrag;
      bfrag.h[0] = *reinterpret_cast<const v8h*>(wp + kk + ks + g * 8);
      bfrag.h[1] = *reinterpret_cast<const v8h*>(wp + kk + ks + 16 + g * 8);
#pragma unroll
      for (int t = 0; t < 4; ++t) {
        F16x16 afrag;
        const _Float16* ar = abuf + (t * 16 + r) * AROW + ks;
        afrag.h[0] = *reinterpret_cast<const v8h*>(ar + g * 8);
        afrag.h[1] = *reinterpret_cast<const v8h*>(ar + 16 + g * 8);
        acc[t] = __builtin_amdgcn_wmma_f32_16x16x32_f16(
            false, afrag.v, false, bfrag.v, (short)0, acc[t], false, false);
      }
    }
    __syncthreads();                                       // protect buffer for re-staging
  }

  const float bn = bias ? bias[col] : 0.0f;
#pragma unroll
  for (int t = 0; t < 4; ++t) {
#pragma unroll
    for (int j = 0; j < 8; ++j) {
      int m = mBase + t * 16 + g * 8 + j;    // D layout: lane group selects M half
      if (m < M) {
        float y = acc[t][j] + bn;
        if (mode == 0) {
          outF[(size_t)m * N + col] = y;
        } else {                              // tanh-approx GELU, f16 out
          float u  = 0.7978845608028654f * (y + 0.044715f * y * y * y);
          float gv = 0.5f * y * (1.0f + tanhf(u));
          outH[(size_t)m * N + col] = (_Float16)gv;
        }
      }
    }
  }
}

// ---------------- embedding + LN ----------------
__global__ void __launch_bounds__(256) embed_ln_k(const int* __restrict__ ids,
                                                  const float* __restrict__ emb,
                                                  const float* __restrict__ pos,
                                                  const float* __restrict__ gg,
                                                  const float* __restrict__ bb,
                                                  float* __restrict__ xf,
                                                  _Float16* __restrict__ xh) {
  int row = blockIdx.x;
  int s   = row % S_;
  int id  = ids[row];
  int tid = threadIdx.x;
  __shared__ float red[256];
  float vals[3];
  float lsum = 0.f;
#pragma unroll
  for (int i = 0; i < 3; ++i) {
    int d = tid + i * 256;
    float v = emb[(size_t)id * D_ + d] + pos[(size_t)s * D_ + d];
    vals[i] = v; lsum += v;
  }
  red[tid] = lsum; __syncthreads();
  for (int st = 128; st > 0; st >>= 1) { if (tid < st) red[tid] += red[tid + st]; __syncthreads(); }
  float mean = red[0] * (1.0f / D_);
  __syncthreads();
  float lvar = 0.f;
#pragma unroll
  for (int i = 0; i < 3; ++i) { float dv = vals[i] - mean; lvar += dv * dv; }
  red[tid] = lvar; __syncthreads();
  for (int st = 128; st > 0; st >>= 1) { if (tid < st) red[tid] += red[tid + st]; __syncthreads(); }
  float rstd = rsqrtf(red[0] * (1.0f / D_) + 1e-5f);
#pragma unroll
  for (int i = 0; i < 3; ++i) {
    int d = tid + i * 256;
    float y = (vals[i] - mean) * rstd * gg[d] + bb[d];
    xf[(size_t)row * D_ + d] = y;
    xh[(size_t)row * D_ + d] = (_Float16)y;
  }
}

// ---------------- residual + LN (in-place on xf, also writes f16) ----------------
__global__ void __launch_bounds__(256) add_ln_k(float* __restrict__ xf,
                                                const float* __restrict__ delta,
                                                const float* __restrict__ gg,
                                                const float* __restrict__ bb,
                                                _Float16* __restrict__ xh) {
  int row = blockIdx.x;
  int tid = threadIdx.x;
  __shared__ float red[256];
  float vals[3];
  float lsum = 0.f;
#pragma unroll
  for (int i = 0; i < 3; ++i) {
    int d = tid + i * 256;
    float v = xf[(size_t)row * D_ + d] + delta[(size_t)row * D_ + d];
    vals[i] = v; lsum += v;
  }
  red[tid] = lsum; __syncthreads();
  for (int st = 128; st > 0; st >>= 1) { if (tid < st) red[tid] += red[tid + st]; __syncthreads(); }
  float mean = red[0] * (1.0f / D_);
  __syncthreads();
  float lvar = 0.f;
#pragma unroll
  for (int i = 0; i < 3; ++i) { float dv = vals[i] - mean; lvar += dv * dv; }
  red[tid] = lvar; __syncthreads();
  for (int st = 128; st > 0; st >>= 1) { if (tid < st) red[tid] += red[tid + st]; __syncthreads(); }
  float rstd = rsqrtf(red[0] * (1.0f / D_) + 1e-5f);
#pragma unroll
  for (int i = 0; i < 3; ++i) {
    int d = tid + i * 256;
    float y = (vals[i] - mean) * rstd * gg[d] + bb[d];
    xf[(size_t)row * D_ + d] = y;
    xh[(size_t)row * D_ + d] = (_Float16)y;
  }
}

// ---------------- local sliding-chunk attention + global column ----------------
__global__ void __launch_bounds__(192) local_attn_k(const float* __restrict__ qf,
                                                    const float* __restrict__ kf,
                                                    const float* __restrict__ vf,
                                                    const int* __restrict__ mask,
                                                    float* __restrict__ outp) {
  int blk = blockIdx.x;                 // b*H*C + h*C + c
  int c = blk % C_;
  int h = (blk / C_) % H_;
  int b = blk / (C_ * H_);
  int w = threadIdx.x;
  if (w >= W_) return;
  int s = c * W_ + w;
  const size_t hoff = (size_t)h * DH_;
  const float scale = 0.125f;           // 1/sqrt(64)

  float4 qv[16];
  const float4* qp = reinterpret_cast<const float4*>(qf + (size_t)(b * S_ + s) * D_ + hoff);
#pragma unroll
  for (int i = 0; i < 16; ++i) {
    float4 q = qp[i];
    qv[i] = make_float4(q.x * scale, q.y * scale, q.z * scale, q.w * scale);
  }

  float mx = -3.0e38f, den = 0.f;
  float acc[64];
#pragma unroll
  for (int d = 0; d < 64; ++d) acc[d] = 0.f;

  for (int j = -1; j < 3 * W_; ++j) {
    int kpos = (j < 0) ? 0 : (c - 1) * W_ + j;       // j==-1: global column (token 0)
    if (j >= 0 && (kpos < 0 || kpos >= S_ || kpos == 0)) continue;
    if (mask[b * S_ + kpos] == 0) continue;
    const float4* kp = reinterpret_cast<const float4*>(kf + (size_t)(b * S_ + kpos) * D_ + hoff);
    float dot = 0.f;
#pragma unroll
    for (int i = 0; i < 16; ++i) {
      float4 kv = kp[i];
      dot += qv[i].x * kv.x + qv[i].y * kv.y + qv[i].z * kv.z + qv[i].w * kv.w;
    }
    float e;
    if (dot > mx) {
      float corr = __expf(mx - dot);
      den *= corr;
#pragma unroll
      for (int d = 0; d < 64; ++d) acc[d] *= corr;
      mx = dot; e = 1.0f;
    } else {
      e = __expf(dot - mx);
    }
    den += e;
    const float4* vp = reinterpret_cast<const float4*>(vf + (size_t)(b * S_ + kpos) * D_ + hoff);
#pragma unroll
    for (int i = 0; i < 16; ++i) {
      float4 vv = vp[i];
      acc[4 * i + 0] += e * vv.x; acc[4 * i + 1] += e * vv.y;
      acc[4 * i + 2] += e * vv.z; acc[4 * i + 3] += e * vv.w;
    }
  }
  float inv = 1.0f / den;
  float* op = outp + (size_t)(b * S_ + s) * D_ + hoff;
#pragma unroll
  for (int d = 0; d < 64; ++d) op[d] = acc[d] * inv;
}

// ---------------- token-0 global attention (overwrites row s=0) ----------------
__global__ void __launch_bounds__(256) global_attn_k(const float* __restrict__ qg0,
                                                     const float* __restrict__ kgf,
                                                     const float* __restrict__ vgf,
                                                     const int* __restrict__ mask,
                                                     float* __restrict__ outp) {
  int h = blockIdx.x % H_;
  int b = blockIdx.x / H_;
  int tid = threadIdx.x;
  const size_t hoff = (size_t)h * DH_;
  __shared__ float sc[S_];
  __shared__ float red[256];
  __shared__ float qsh[DH_];
  if (tid < DH_) qsh[tid] = qg0[(size_t)b * D_ + hoff + tid];   // already scaled
  __syncthreads();

  float lmax = -3.0e38f;
  for (int s = tid; s < S_; s += 256) {
    float dot;
    if (mask[b * S_ + s] != 0) {
      const float* kp = kgf + (size_t)(b * S_ + s) * D_ + hoff;
      dot = 0.f;
#pragma unroll
      for (int d = 0; d < DH_; ++d) dot += qsh[d] * kp[d];
    } else {
      dot = -1.0e9f;
    }
    sc[s] = dot;
    lmax = fmaxf(lmax, dot);
  }
  red[tid] = lmax; __syncthreads();
  for (int st = 128; st > 0; st >>= 1) { if (tid < st) red[tid] = fmaxf(red[tid], red[tid + st]); __syncthreads(); }
  float mx = red[0];
  __syncthreads();
  float lsum = 0.f;
  for (int s = tid; s < S_; s += 256) { float e = __expf(sc[s] - mx); sc[s] = e; lsum += e; }
  red[tid] = lsum; __syncthreads();
  for (int st = 128; st > 0; st >>= 1) { if (tid < st) red[tid] += red[tid + st]; __syncthreads(); }
  float den = red[0];
  __syncthreads();
  if (tid < DH_) {
    float a = 0.f;
    for (int s = 0; s < S_; ++s) a += sc[s] * vgf[(size_t)(b * S_ + s) * D_ + hoff + tid];
    outp[(size_t)(b * S_) * D_ + hoff + tid] = a / den;
  }
}

// ---------------- token-0 global-Q projection: qg0 = (x0 @ Wqg + bqg) * scale ----
__global__ void __launch_bounds__(256) qg_proj_k(const float* __restrict__ xf,
                                                 const float* __restrict__ Wqg,
                                                 const float* __restrict__ bqg,
                                                 float* __restrict__ qg0) {
  int idx = blockIdx.x * 256 + threadIdx.x;
  if (idx >= B_ * D_) return;
  int b = idx / D_, j = idx - b * D_;
  const float* xr = xf + (size_t)(b * S_) * D_;
  float a = bqg[j];
  for (int d = 0; d < D_; ++d) a += xr[d] * Wqg[(size_t)d * D_ + j];
  qg0[idx] = a * 0.125f;
}

// ---------------- copy CLS row ----------------
__global__ void __launch_bounds__(256) copy_cls_k(const float* __restrict__ xf,
                                                  float* __restrict__ out) {
  int idx = blockIdx.x * 256 + threadIdx.x;
  if (idx >= B_ * D_) return;
  int b = idx / D_, d = idx - b * D_;
  out[idx] = xf[(size_t)(b * S_) * D_ + d];
}

// =====================================================================
extern "C" void kernel_launch(void* const* d_in, const int* in_sizes, int n_in,
                              void* d_out, int out_size, void* d_ws, size_t ws_size,
                              hipStream_t stream) {
  const int* src_ids  = (const int*)d_in[0];
  const int* src_mask = (const int*)d_in[1];
  const int* tgt_ids  = (const int*)d_in[2];
  const int* tgt_mask = (const int*)d_in[3];
  int p = 4;
  const float* emb      = (const float*)d_in[p++];
  const float* pos      = (const float*)d_in[p++];
  const float* emb_ln_g = (const float*)d_in[p++];
  const float* emb_ln_b = (const float*)d_in[p++];
  const float* Wq  = (const float*)d_in[p++]; const float* bq  = (const float*)d_in[p++];
  const float* Wk  = (const float*)d_in[p++]; const float* bk  = (const float*)d_in[p++];
  const float* Wv  = (const float*)d_in[p++]; const float* bv  = (const float*)d_in[p++];
  const float* Wqg = (const float*)d_in[p++]; const float* bqg = (const float*)d_in[p++];
  const float* Wkg = (const float*)d_in[p++]; const float* bkg = (const float*)d_in[p++];
  const float* Wvg = (const float*)d_in[p++]; const float* bvg = (const float*)d_in[p++];
  const float* Wo  = (const float*)d_in[p++]; const float* bo  = (const float*)d_in[p++];
  const float* ln1_g = (const float*)d_in[p++]; const float* ln1_b = (const float*)d_in[p++];
  const float* W1  = (const float*)d_in[p++]; const float* b1  = (const float*)d_in[p++];
  const float* W2  = (const float*)d_in[p++]; const float* b2  = (const float*)d_in[p++];
  const float* ln2_g = (const float*)d_in[p++]; const float* ln2_b = (const float*)d_in[p++];

  // ---- workspace carve ----
  char* ws = (char*)d_ws;
  size_t off = 0;
  auto carve = [&](size_t bytes) -> void* {
    void* r = ws + off;
    off = (off + bytes + 255) & ~(size_t)255;
    return r;
  };
  float*    xf    = (float*)   carve((size_t)M_ * D_ * 4);
  _Float16* xh    = (_Float16*)carve((size_t)M_ * D_ * 2);
  float*    qf    = (float*)   carve((size_t)M_ * D_ * 4);
  float*    kf    = (float*)   carve((size_t)M_ * D_ * 4);
  float*    vf    = (float*)   carve((size_t)M_ * D_ * 4);
  float*    attnf = (float*)   carve((size_t)M_ * D_ * 4);
  float*    tmpf  = (float*)   carve((size_t)M_ * D_ * 4);
  _Float16* ah    = (_Float16*)carve((size_t)M_ * D_ * 2);
  _Float16* hh    = (_Float16*)carve((size_t)M_ * FF_ * 2);
  _Float16* wt    = (_Float16*)carve((size_t)FF_ * D_ * 2);
  float*    qg0   = (float*)   carve((size_t)B_ * D_ * 4);
  float*    kgf   = qf;   // qf/kf dead after local_attn (stream-ordered reuse)
  float*    vgf   = kf;

  float* out = (float*)d_out;

  auto cvt = [&](const float* Wsrc, int K, int N) {
    int tot = K * N;
    hipLaunchKernelGGL(cvt_wT_k, dim3((tot + 255) / 256), dim3(256), 0, stream, Wsrc, wt, K, N);
  };
  auto gemm = [&](const _Float16* X, const float* bias, float* oF, _Float16* oH,
                  int M, int N, int K, int mode) {
    dim3 g((M + 63) / 64, N / 128);
    hipLaunchKernelGGL(gemm_wmma_k, g, dim3(256), 0, stream, X, wt, bias, oF, oH, M, N, K, mode);
  };

  for (int pass = 0; pass < 2; ++pass) {
    const int* ids = pass ? tgt_ids : src_ids;
    const int* msk = pass ? tgt_mask : src_mask;

    hipLaunchKernelGGL(embed_ln_k, dim3(M_), dim3(256), 0, stream,
                       ids, emb, pos, emb_ln_g, emb_ln_b, xf, xh);

    for (int l = 0; l < L_; ++l) {
      const size_t wo = (size_t)l * D_ * D_;
      const size_t bo_ = (size_t)l * D_;

      cvt(Wq + wo, D_, D_);  gemm(xh, bq + bo_, qf, nullptr, M_, D_, D_, 0);
      cvt(Wk + wo, D_, D_);  gemm(xh, bk + bo_, kf, nullptr, M_, D_, D_, 0);
      cvt(Wv + wo, D_, D_);  gemm(xh, bv + bo_, vf, nullptr, M_, D_, D_, 0);

      hipLaunchKernelGGL(local_attn_k, dim3(B_ * H_ * C_), dim3(192), 0, stream,
                         qf, kf, vf, msk, attnf);

      hipLaunchKernelGGL(qg_proj_k, dim3((B_ * D_ + 255) / 256), dim3(256), 0, stream,
                         xf, Wqg + wo, bqg + bo_, qg0);
      cvt(Wkg + wo, D_, D_); gemm(xh, bkg + bo_, kgf, nullptr, M_, D_, D_, 0);
      cvt(Wvg + wo, D_, D_); gemm(xh, bvg + bo_, vgf, nullptr, M_, D_, D_, 0);
      hipLaunchKernelGGL(global_attn_k, dim3(B_ * H_), dim3(256), 0, stream,
                         qg0, kgf, vgf, msk, attnf);

      hipLaunchKernelGGL(cvt_h_k, dim3((M_ * D_ + 255) / 256), dim3(256), 0, stream,
                         attnf, ah, M_ * D_);
      cvt(Wo + wo, D_, D_);  gemm(ah, bo + bo_, tmpf, nullptr, M_, D_, D_, 0);
      hipLaunchKernelGGL(add_ln_k, dim3(M_), dim3(256), 0, stream,
                         xf, tmpf, ln1_g + bo_, ln1_b + bo_, xh);

      cvt(W1 + (size_t)l * D_ * FF_, D_, FF_);
      gemm(xh, b1 + (size_t)l * FF_, nullptr, hh, M_, FF_, D_, 1);   // GELU -> f16
      cvt(W2 + (size_t)l * FF_ * D_, FF_, D_);
      gemm(hh, b2 + bo_, tmpf, nullptr, M_, D_, FF_, 0);
      hipLaunchKernelGGL(add_ln_k, dim3(M_), dim3(256), 0, stream,
                         xf, tmpf, ln2_g + bo_, ln2_b + bo_, xh);
    }

    hipLaunchKernelGGL(copy_cls_k, dim3((B_ * D_ + 255) / 256), dim3(256), 0, stream,
                       xf, out + (size_t)pass * B_ * D_);
  }
  (void)in_sizes; (void)n_in; (void)out_size; (void)ws_size;
}
